// SupConLoss_34737695490547
// MI455X (gfx1250) — matrix operand order repaired and checked by reading, not compiled, not run on hardware
//
#include <hip/hip_runtime.h>
#include <stddef.h>

#define BSZ    2048
#define NVIEW  2
#define NTOT   (BSZ*NVIEW)          // 4096
#define DIM    512
#define NCLS   64
#define QRT    1024                 // N processed per LDS pass (4 passes)
#define SSTR   1028                 // padded LDS row stride (1028 % 64 == 4 -> conflict-free stores)
#define ASTR   520                  // A-panel LDS row stride in bf16 (260 dwords; 260%64==4)

typedef __attribute__((ext_vector_type(8)))  __bf16 bf16x8;
typedef __attribute__((ext_vector_type(16))) __bf16 bf16x16;
typedef __attribute__((ext_vector_type(8)))  float  v8f;
typedef int v4i __attribute__((vector_size(16)));   // matches builtin's expected pointee

#if defined(__has_builtin)
# if __has_builtin(__builtin_amdgcn_global_load_async_to_lds_b128)
#  define HAVE_ASYNC_LDS 1
# endif
#endif
#define AS1 __attribute__((address_space(1)))
#define AS3 __attribute__((address_space(3)))

// ---------------- helpers ----------------
__device__ __forceinline__ float wred_sum(float v){
#pragma unroll
  for (int m = 16; m > 0; m >>= 1) v += __shfl_xor(v, m, 32);
  return v;
}
__device__ __forceinline__ float wred_max(float v){
#pragma unroll
  for (int m = 16; m > 0; m >>= 1) v = fmaxf(v, __shfl_xor(v, m, 32));
  return v;
}
__device__ __forceinline__ unsigned short f2bf_rne(float x){
  unsigned u = __float_as_uint(x);
  unsigned r = u + 0x7FFFu + ((u >> 16) & 1u);
  return (unsigned short)(r >> 16);
}
__device__ __forceinline__ float bf2f(unsigned short h){
  return __uint_as_float(((unsigned)h) << 16);
}

// A fragment from LDS panel: 16x32 bf16, lane group g: K = 16g..16g+15 (contiguous)
__device__ __forceinline__ bf16x16 fragA_lds(const __bf16* pan, int rowl, int k, int g){
  const __bf16* p = pan + rowl * ASTR + k + 16*g;
  bf16x8 a = *(const bf16x8*)p;          // ds_load_b128
  bf16x8 b = *(const bf16x8*)(p + 8);    // ds_load_b128
  return __builtin_shufflevector(a, b, 0,1,2,3,4,5,6,7,8,9,10,11,12,13,14,15);
}
// B fragment from global: 32x16 bf16, lane group g: K = 16g..16g+15 (contiguous)
__device__ __forceinline__ bf16x16 fragB(const __bf16* M, int row, int k, int g){
  const __bf16* p = M + (size_t)row * DIM + k + 16*g;
  return *(const bf16x16*)p;
}
__device__ __forceinline__ v8f wmma_bf16(bf16x16 a, bf16x16 b, v8f c){
  return __builtin_amdgcn_wmma_f32_16x16x32_bf16(false, a, false, b, (short)0, c,
                                                 false, false);
}

// ---------------- K1: L2-normalize rows (view-major) + bf16 hi/lo split ----------------
__global__ __launch_bounds__(128)
void k_normalize_split(const float* __restrict__ feats,
                       unsigned short* __restrict__ chi,
                       unsigned short* __restrict__ clo){
  int r = blockIdx.x;                    // contrast row r = v*BSZ + b
  int v = r >> 11, b = r & (BSZ - 1);
  const float* src = feats + (size_t)(b * NVIEW + v) * DIM;
  float ss = 0.f;
  for (int k = threadIdx.x; k < DIM; k += 128){ float x = src[k]; ss += x * x; }
  ss = wred_sum(ss);
  __shared__ float part[4];
  if ((threadIdx.x & 31) == 0) part[threadIdx.x >> 5] = ss;
  __syncthreads();
  float inv = 1.f / fmaxf(sqrtf(part[0] + part[1] + part[2] + part[3]), 1e-12f);
  size_t base = (size_t)r * DIM;
  for (int k = threadIdx.x; k < DIM; k += 128){
    float y = src[k] * inv;
    unsigned short h = f2bf_rne(y);
    unsigned short l = f2bf_rne(y - bf2f(h));
    chi[base + k] = h;
    clo[base + k] = l;
  }
}

// ---------------- K2: fused 16-row strip: WMMA GEMM (split-bf16) + supcon row math ----------------
// launch_bounds(256,4): <=256 VGPRs (no s_set_vgpr_msb), 2 workgroups / WGP.
__global__ __launch_bounds__(256, 4)
void k_supcon_fused(const unsigned short* __restrict__ chi_u,
                    const unsigned short* __restrict__ clo_u,
                    const int* __restrict__ labels,
                    float* __restrict__ rowloss){
  const __bf16* Chi = (const __bf16*)chi_u;
  const __bf16* Clo = (const __bf16*)clo_u;

  __shared__ float        Stile[16 * SSTR];       // 64.25 KB : 16-row x 1024-col dot strip
  __shared__ unsigned int Apan [2 * 16 * 260];    // 32.5 KB : k-invariant A panel (Chi|Clo), padded
  __shared__ float        posv [16][64];          // 4 KB    : per-row candidate positives
  __shared__ float        spart[16];              // partner similarity

  const int m0   = blockIdx.x * 16;
  const int tid  = threadIdx.x;
  const int lane = tid & 31;
  const int wave = tid >> 5;             // 8 waves
  const int g    = lane >> 4;
  const int l16  = lane & 15;

  const float invT = 1.0f / 0.07f;
  const float twoT = 2.0f / 0.07f;

  float rmax[2] = { -__builtin_inff(), -__builtin_inff() };
  float rsum[2] = { 0.f, 0.f };
  if (lane == 0){ spart[2*wave] = 0.f; spart[2*wave + 1] = 0.f; }

  // ---- stage A panel (16 rows x 512 of Chi and Clo) into LDS: async-to-LDS if available ----
  {
#pragma unroll
    for (int q = 0; q < 8; ++q){
      int c    = tid + 256 * q;            // 2048 chunks of 16B
      int mat  = c >> 10;                  // 0 = Chi, 1 = Clo
      int rowc = (c >> 6) & 15;
      int dw4  = (c & 63) * 4;             // dword index within row (0..252)
      const unsigned short* gp = (mat ? clo_u : chi_u)
                                 + (size_t)(m0 + rowc) * DIM + (size_t)dw4 * 2;
      unsigned int* lp = &Apan[(mat * 16 + rowc) * 260 + dw4];
#if defined(HAVE_ASYNC_LDS)
      __builtin_amdgcn_global_load_async_to_lds_b128((AS1 v4i*)gp, (AS3 v4i*)lp, 0, 0);
#else
      uint4 v = *(const uint4*)gp;
      *(uint4*)lp = v;
#endif
    }
#if defined(HAVE_ASYNC_LDS)
# if __has_builtin(__builtin_amdgcn_s_wait_asynccnt)
    __builtin_amdgcn_s_wait_asynccnt(0);
# else
    asm volatile("s_wait_asynccnt 0x0" ::: "memory");
# endif
#endif
  }
  __syncthreads();

  const __bf16* ApanH = (const __bf16*)Apan;             // Chi rows 0..15
  const __bf16* ApanL = ApanH + 16 * ASTR;               // Clo rows 0..15

  for (int h = 0; h < 4; ++h){
    // ---- GEMM phase: 64 N-tiles this quarter; wave does groups of 4 tiles sharing A ----
    for (int p = 0; p < 2; ++p){
      int nt[4], nrow[4];
      v8f acc[4];
#pragma unroll
      for (int j = 0; j < 4; ++j){
        nt[j]   = wave + 8 * (4 * p + j);        // 0..63
        nrow[j] = h * QRT + nt[j] * 16 + l16;
        acc[j]  = (v8f){};
        __builtin_prefetch(Chi + (size_t)nrow[j] * DIM, 0, 1);  // global_prefetch_b8
        __builtin_prefetch(Clo + (size_t)nrow[j] * DIM, 0, 1);
      }
      for (int k = 0; k < DIM; k += 32){
        bf16x16 ah = fragA_lds(ApanH, l16, k, g);
        bf16x16 al = fragA_lds(ApanL, l16, k, g);
#pragma unroll
        for (int j = 0; j < 4; ++j){
          bf16x16 bh = fragB(Chi, nrow[j], k, g);
          bf16x16 bl = fragB(Clo, nrow[j], k, g);
          // split-fp32 product: hi*hi + hi*lo + lo*hi  (lo*lo negligible)
          acc[j] = wmma_bf16(ah, bh, acc[j]);
          acc[j] = wmma_bf16(ah, bl, acc[j]);
          acc[j] = wmma_bf16(al, bh, acc[j]);
        }
      }
      // C/D layout: VGPR r -> M = r + 8*g, N = lane&15
#pragma unroll
      for (int j = 0; j < 4; ++j){
        const int c = nt[j] * 16 + l16;
#pragma unroll
        for (int r = 0; r < 8; ++r){
          Stile[(8 * g + r) * SSTR + c] = acc[j][r];
        }
      }
    }
    __syncthreads();

    // ---- row-wise partial pass over this quarter (wave owns rows 2w, 2w+1) ----
    for (int rr = 0; rr < 2; ++rr){
      const int lr = 2 * wave + rr;
      const int i  = m0 + lr;
      const float* Srow = &Stile[lr * SSTR];

      float mx = -__builtin_inff();
      for (int c = lane; c < QRT; c += 32) mx = fmaxf(mx, Srow[c]);
      mx = wred_max(mx) * invT;

      float sm = 0.f;
      for (int c = lane; c < QRT; c += 32){
        int j = h * QRT + c;
        sm += (j == i) ? 0.f : __expf(Srow[c] * invT - mx);
      }
      sm = wred_sum(sm);

      float nm = fmaxf(rmax[rr], mx);
      rsum[rr] = rsum[rr] * __expf(rmax[rr] - nm) + sm * __expf(mx - nm);
      rmax[rr] = nm;

      // positives at columns j = ell + 64*t (balanced labels); 16 per quarter (lanes 0..15)
      const int ell = labels[i & (BSZ - 1)];
      if (lane < 16){
        const int lcol = ell + 64 * lane;          // within quarter
        const int j    = h * QRT + lcol;
        const int partner = i ^ BSZ;
        const float s  = Srow[lcol] * invT;
        float val = s + twoT;
        if (j == i || j == partner) val = -1e30f;  // sentinels (self + aug pair)
        if (j == partner) spart[lr] = s;
        posv[lr][h * 16 + lane] = val;
      }
    }
    __syncthreads();   // protect Stile before next quarter overwrites it
  }

  // ---- median threshold (ranks 32/33 of 64 incl. two sentinels == ranks 30/31 of 62) ----
  for (int rr = 0; rr < 2; ++rr){
    const int lr = 2 * wave + rr;
    const int i  = m0 + lr;
    const float v0 = posv[lr][lane];
    const float v1 = posv[lr][lane + 32];
    int r0 = 0, r1 = 0;
#pragma unroll 8
    for (int k = 0; k < 64; ++k){
      float u = posv[lr][k];
      r0 += (u < v0) || (u == v0 && k < lane);
      r1 += (u < v1) || (u == v1 && k < lane + 32);
    }
    float a32 = ((r0 == 32) ? v0 : 0.f) + ((r1 == 32) ? v1 : 0.f);
    float a33 = ((r0 == 33) ? v0 : 0.f) + ((r1 == 33) ? v1 : 0.f);
    a32 = wred_sum(a32);
    a33 = wred_sum(a33);
    const float thr = 0.5f * (a32 + a33);

    float cnt = 0.f, ssum = 0.f;
    if (v0 > thr){ cnt += 1.f; ssum += v0 - (2.0f / 0.07f); }
    if (v1 > thr){ cnt += 1.f; ssum += v1 - (2.0f / 0.07f); }
    cnt  = wred_sum(cnt);
    ssum = wred_sum(ssum);

    if (lane == 0){
      const float P    = cnt + 1.f;                 // + augmentation partner
      const float tot  = ssum + spart[lr];
      const float logZ = rmax[rr] + __logf(rsum[rr]);
      rowloss[i] = -((tot - P * logZ) / P);         // T/base_T == 1
    }
  }
}

// ---------------- K3: mean over 4096 row losses ----------------
__global__ __launch_bounds__(256)
void k_finalize(const float* __restrict__ rowloss, float* __restrict__ out){
  __shared__ float part[8];
  float s = 0.f;
  for (int k = threadIdx.x; k < NTOT; k += 256) s += rowloss[k];
  s = wred_sum(s);
  if ((threadIdx.x & 31) == 0) part[threadIdx.x >> 5] = s;
  __syncthreads();
  if (threadIdx.x == 0){
    float t = 0.f;
    for (int w = 0; w < 8; ++w) t += part[w];
    out[0] = t * (1.0f / (float)NTOT);
  }
}

extern "C" void kernel_launch(void* const* d_in, const int* in_sizes, int n_in,
                              void* d_out, int out_size, void* d_ws, size_t ws_size,
                              hipStream_t stream){
  (void)in_sizes; (void)n_in; (void)out_size; (void)ws_size;
  const float* feats  = (const float*)d_in[0];
  const int*   labels = (const int*)d_in[1];

  unsigned short* chi = (unsigned short*)d_ws;                    // 4096*512 bf16 (4 MB)
  unsigned short* clo = chi + (size_t)NTOT * DIM;                 // 4 MB
  float* rowloss = (float*)(clo + (size_t)NTOT * DIM);            // 16 KB

  k_normalize_split<<<NTOT, 128, 0, stream>>>(feats, chi, clo);
  k_supcon_fused<<<NTOT / 16, 256, 0, stream>>>(chi, clo, labels, rowloss);
  k_finalize<<<1, 256, 0, stream>>>(rowloss, (float*)d_out);
}
